// HierarchicalBG_89498528514408
// MI455X (gfx1250) — matrix operand order repaired and checked by reading, not compiled, not run on hardware
//
#include <hip/hip_runtime.h>
#include <hip/hip_bf16.h>

#define BPTS 128                    // points per block (4 wave32)
#define PI_F 3.14159265358979323846f

typedef __attribute__((ext_vector_type(16))) _Float16 v16h;
typedef __attribute__((ext_vector_type(8)))  _Float16 v8h;
typedef __attribute__((ext_vector_type(8)))  float    v8f;

union V16 { v16h v; v8h h[2]; };

__device__ __forceinline__ float softplus_f(float x) {
    // log(1+exp(x)) = max(x,0) + log1p(exp(-|x|))
    return fmaxf(x, 0.0f) + log1pf(__expf(-fabsf(x)));
}

// Bilinear gather of 16 channels from a (16,H,W) grid; weights zeroed for
// out-of-range taps (matches reference's validity masking). NT = bypass
// temporal caching for the 512MB level-1 grid (larger than the 192MB L2).
template<int H, int W, bool NT>
__device__ __forceinline__ void sample_grid(const float* __restrict__ mat,
                                            float gx, float gy, float scale,
                                            float acc[16]) {
    const float px  = ((gx + 1.0f) * (float)W - 1.0f) * 0.5f;
    const float py  = ((gy + 1.0f) * (float)H - 1.0f) * 0.5f;
    const float x0f = floorf(px), y0f = floorf(py);
    const float wx1 = px - x0f,   wy1 = py - y0f;
    const float wx0 = 1.0f - wx1, wy0 = 1.0f - wy1;
    const int x0 = (int)x0f, y0 = (int)y0f;

    const int   xs[2]  = { x0, x0 + 1 };
    const int   ys[2]  = { y0, y0 + 1 };
    const float wxs[2] = { wx0, wx1 };
    const float wys[2] = { wy0, wy1 };

    long  bases[4];
    float wts[4];
#pragma unroll
    for (int j = 0; j < 2; ++j)
#pragma unroll
        for (int i = 0; i < 2; ++i) {
            const int xi = xs[i], yi = ys[j];
            const bool valid = (xi >= 0) && (xi < W) && (yi >= 0) && (yi < H);
            const int xc = min(max(xi, 0), W - 1);
            const int yc = min(max(yi, 0), H - 1);
            bases[j * 2 + i] = (long)yc * W + xc;
            wts[j * 2 + i]   = valid ? (wxs[i] * wys[j] * scale) : 0.0f;
        }

#pragma unroll
    for (int r = 0; r < 16; ++r) {
        const float* p = mat + (long)r * ((long)H * W);
#pragma unroll
        for (int t = 0; t < 4; ++t) {
            const float v = NT ? __builtin_nontemporal_load(p + bases[t])
                               : p[bases[t]];
            acc[r] += wts[t] * v;
        }
    }
}

__global__ __launch_bounds__(BPTS, 1)
void hbg_fused_kernel(const float* __restrict__ viewdirs,
                      const float* __restrict__ mat0,   // (16,512,1024)
                      const float* __restrict__ mat1,   // (16,2048,4096)
                      const float* __restrict__ w1,     // (16,128)
                      const float* __restrict__ w2,     // (128,3)
                      float* __restrict__ out,          // (B,3)
                      int B) {
    // All WMMA operands are staged zero-PADDED in LDS so fragment loads are
    // unconditional (per-lane address cndmask instead of saveexec branches):
    __shared__ __align__(16) _Float16 sEmb[BPTS][32];   // emb rows, K padded 16->32
    __shared__ __align__(16) _Float16 sW1P[128][32];    // w1^T,     K padded 16->32
    __shared__ __align__(16) _Float16 sW2T[16][128];    // w2^T,     N padded 3->16
    __shared__ __align__(16) _Float16 sH[BPTS / 32][16][128]; // per-wave h tile

    const int tid  = threadIdx.x;
    const int lane = tid & 31;
    const int wv   = tid >> 5;
    const long pt  = (long)blockIdx.x * BPTS + tid;
    const long ptc = (pt < (long)B) ? pt : (long)B - 1;

    // ---- stage weights (f16, zero-padded) ----
#pragma unroll
    for (int k = 0; k < 16; ++k)
        sW1P[tid][k] = (_Float16)w1[k * 128 + tid];
#pragma unroll
    for (int k = 16; k < 32; ++k)
        sW1P[tid][k] = (_Float16)0.0f;
    if (tid < 16) {
        for (int K = 0; K < 128; ++K)
            sW2T[tid][K] = (tid < 3) ? (_Float16)w2[K * 3 + tid] : (_Float16)0.0f;
    }

    // ---- phase 1: per-point equirect unwrap + bilinear gather (the bound) ----
    float acc[16];
#pragma unroll
    for (int r = 0; r < 16; ++r) acc[r] = 0.0f;
    {
        const float vx = viewdirs[ptc * 3 + 0];
        const float vy = viewdirs[ptc * 3 + 1];
        const float vz = viewdirs[ptc * 3 + 2];
        const float inv = 1.0f / (sqrtf(vx * vx + vy * vy + vz * vz) + 1e-8f);
        const float dz = vz * inv;
        const float gx = atan2f(vx * inv, vy * inv) * (1.0f / PI_F);
        const float gy = acosf(fminf(fmaxf(dz, -1.0f), 1.0f)) * (2.0f / PI_F) - 1.0f;
        sample_grid<512, 1024, false>(mat0, gx, gy, 1.0f, acc);   // 32MB: L2-resident
        sample_grid<2048, 4096, true>(mat1, gx, gy, 0.5f, acc);   // 512MB: non-temporal
    }
#pragma unroll
    for (int r = 0; r < 16; ++r) sEmb[tid][r] = (_Float16)acc[r];
#pragma unroll
    for (int r = 16; r < 32; ++r) sEmb[tid][r] = (_Float16)0.0f;

    __syncthreads();

    // ---- phase 2: WMMA MLP. Each wave owns rows [32*wv, 32*wv+32): two 16-row tiles. ----
    const v8f zeroC = {};

#pragma unroll
    for (int t = 0; t < 2; ++t) {
        const int rowbase = wv * 32 + t * 16;      // block-local first point of tile
        const int mrow    = rowbase + (lane & 15); // this lane's matrix row

        // A fragment (16x32 f16, K zero-padded 16->32):
        //  lanes 0-15 : halves 0-7 = K0..7,  halves 8-15 = K16..23 (pad)
        //  lanes 16-31: halves 0-7 = K8..15, halves 8-15 = K24..31 (pad)
        const _Float16* ap = &sEmb[mrow][(lane < 16) ? 0 : 8];
        V16 a;
        a.h[0] = *(const v8h*)(ap);
        a.h[1] = *(const v8h*)(ap + 16);

        // GEMM1: emb(16x16) @ w1(16x128) over 8 N-tiles, relu -> sH (f16)
#pragma unroll
        for (int n = 0; n < 8; ++n) {
            // B fragment (32x16): lanes 0-15 carry K0..7,K16..23; lanes 16-31 K8..15,K24..31
            const _Float16* bp = &sW1P[n * 16 + (lane & 15)][(lane < 16) ? 0 : 8];
            V16 b;
            b.h[0] = *(const v8h*)(bp);
            b.h[1] = *(const v8h*)(bp + 16);

            v8f d = __builtin_amdgcn_wmma_f32_16x16x32_f16(
                false, a.v, false, b.v, (short)0, zeroC, false, false);

            // D layout: lanes 0-15 -> M=r, lanes 16-31 -> M=r+8; N = lane%16
#pragma unroll
            for (int r = 0; r < 8; ++r) {
                const int row = (lane < 16) ? r : (r + 8);
                sH[wv][row][n * 16 + (lane & 15)] = (_Float16)fmaxf(d[r], 0.0f);
            }
        }

        // Same-wave LDS RAW: DS ops are in-order per wave; make the dependency
        // explicit for the compiler and hardware.
        asm volatile("s_wait_dscnt 0" ::: "memory");

        // GEMM2: h(16x128) @ w2p(128x16) as 4 chained K=32 WMMAs
        v8f acc2 = zeroC;
#pragma unroll
        for (int c = 0; c < 4; ++c) {
            const _Float16* hp = &sH[wv][lane & 15][c * 32 + ((lane < 16) ? 0 : 8)];
            V16 a2;
            a2.h[0] = *(const v8h*)(hp);
            a2.h[1] = *(const v8h*)(hp + 16);

            const _Float16* bp2 = &sW2T[lane & 15][c * 32 + ((lane < 16) ? 0 : 8)];
            V16 b2;
            b2.h[0] = *(const v8h*)(bp2);
            b2.h[1] = *(const v8h*)(bp2 + 16);

            acc2 = __builtin_amdgcn_wmma_f32_16x16x32_f16(
                false, a2.v, false, b2.v, (short)0, acc2, false, false);
        }

        // softplus + scatter the 3 valid output columns
        const int nc = lane & 15;
        if (nc < 3) {
#pragma unroll
            for (int r = 0; r < 8; ++r) {
                const int row = (lane < 16) ? r : (r + 8);
                const long p  = (long)blockIdx.x * BPTS + rowbase + row;
                if (p < (long)B)
                    out[p * 3 + nc] = softplus_f(acc2[r]);
            }
        }
    }
}

extern "C" void kernel_launch(void* const* d_in, const int* in_sizes, int n_in,
                              void* d_out, int out_size, void* d_ws, size_t ws_size,
                              hipStream_t stream) {
    const float* viewdirs = (const float*)d_in[0];
    const float* mat0     = (const float*)d_in[1];
    const float* mat1     = (const float*)d_in[2];
    const float* w1       = (const float*)d_in[3];
    const float* w2       = (const float*)d_in[4];
    float*       out      = (float*)d_out;

    const int B    = in_sizes[0] / 3;
    const int grid = (B + BPTS - 1) / BPTS;
    hbg_fused_kernel<<<dim3(grid), dim3(BPTS), 0, stream>>>(
        viewdirs, mat0, mat1, w1, w2, out, B);
}
